// NovelAttentionLM_24154896072720
// MI455X (gfx1250) — matrix operand order repaired and checked by reading, not compiled, not run on hardware
//
#include <hip/hip_runtime.h>
#include <math.h>

// ---------------- model dims ----------------
#define BATCH  2
#define SEQ    2048
#define DIM    512
#define LAYERS 4
#define VOCAB  32000
#define MROWS  (BATCH * SEQ)        // 4096
#define D4     (4 * DIM)            // 2048

// ---------------- GEMM tiling ----------------
#define BM 128
#define BN 128
#define BK 32
#define LDS_STRIDE 80               // 64B of K-data + 16B pad (conflict-free frag reads)
#define LDS_A_BYTES (BM * LDS_STRIDE)       // 10240
#define LDS_HALF    (LDS_A_BYTES * 2)       // 20480: A tile + B tile
#define LDS_TOTAL   (LDS_HALF * 2)          // 40960: double buffered

typedef __bf16 bf16;
typedef __bf16 v8bf  __attribute__((ext_vector_type(8)));
typedef __bf16 v16bf __attribute__((ext_vector_type(16)));
typedef float  v8f   __attribute__((ext_vector_type(8)));

__device__ __forceinline__ bf16 f2bf(float f) {
    union { float f; unsigned u; } v; v.f = f;
    unsigned u = v.u;
    unsigned r = (u + 0x7FFFu + ((u >> 16) & 1u)) >> 16;
    unsigned short s = (unsigned short)r;
    bf16 out;
    __builtin_memcpy(&out, &s, 2);
    return out;
}

// ---------------- embedding + sinusoidal PE ----------------
__global__ void embed_kernel(const int* __restrict__ tokens,
                             const float* __restrict__ emb,
                             float* __restrict__ x) {
    int tid = blockIdx.x * blockDim.x + threadIdx.x;
    const int total = MROWS * DIM;
    if (tid >= total) return;
    int d = tid % DIM;
    int row = tid / DIM;            // row = b*SEQ + s
    int s = row % SEQ;
    int tok = tokens[row];
    int d2 = d & ~1;
    float div = __expf(-(float)d2 * (9.210340371976184f / (float)DIM));
    float ang = (float)s * div;
    float pe = (d & 1) ? __cosf(ang) : __sinf(ang);
    x[tid] = emb[(size_t)tok * DIM + d] + pe;
}

// ---------------- fp32 [K,N] -> bf16 transposed [N,K] ----------------
__global__ void transpose_convert_kernel(const float* __restrict__ W,
                                         bf16* __restrict__ Bt,
                                         int K, int N) {
    int tid = blockIdx.x * blockDim.x + threadIdx.x;
    if (tid >= K * N) return;
    int k = tid / N;
    int n = tid % N;
    Bt[(size_t)n * K + k] = f2bf(W[tid]);
}

// ---------------- LayerNorm over last dim (DIM=512), one row per block ----------------
__global__ void layernorm_kernel(const float* __restrict__ x,
                                 const float* __restrict__ w,
                                 const float* __restrict__ b,
                                 float* __restrict__ h,
                                 bf16* __restrict__ hbf) {
    __shared__ float red[256];
    int row = blockIdx.x;
    int tid = threadIdx.x;
    const float* xr = x + (size_t)row * DIM;
    float v0 = xr[tid];
    float v1 = xr[tid + 256];
    red[tid] = v0 + v1;
    __syncthreads();
    for (int s = 128; s > 0; s >>= 1) {
        if (tid < s) red[tid] += red[tid + s];
        __syncthreads();
    }
    float mu = red[0] * (1.0f / DIM);
    __syncthreads();
    float d0 = v0 - mu, d1 = v1 - mu;
    red[tid] = d0 * d0 + d1 * d1;
    __syncthreads();
    for (int s = 128; s > 0; s >>= 1) {
        if (tid < s) red[tid] += red[tid + s];
        __syncthreads();
    }
    float var = red[0] * (1.0f / DIM);
    float rs = rsqrtf(var + 1e-5f);
    float o0 = d0 * rs * w[tid]       + b[tid];
    float o1 = d1 * rs * w[tid + 256] + b[tid + 256];
    size_t base = (size_t)row * DIM;
    h[base + tid]        = o0;
    h[base + tid + 256]  = o1;
    hbf[base + tid]       = f2bf(o0);
    hbf[base + tid + 256] = f2bf(o1);
}

// ---------------- sequential phase-interference scan, one thread per (b,e) channel ----------------
__global__ void tpi_scan_kernel(const float* __restrict__ h,
                                const float* __restrict__ omega,
                                const float* __restrict__ mag,
                                const float* __restrict__ iscale,
                                float* __restrict__ ctx,
                                bf16* __restrict__ ctxbf) {
    int tid = blockIdx.x * blockDim.x + threadIdx.x;
    if (tid >= BATCH * DIM) return;
    int b = tid / DIM;
    int e = tid % DIM;
    float isc = fabsf(iscale[e]);
    float phi = 0.f, mr = 0.f, mi = 0.f, acc = 0.f;
    for (int s = 0; s < SEQ; ++s) {
        size_t base = ((size_t)(b * SEQ + s)) * DIM + e;
        float om = omega[base];
        float mg = mag[base];
        float hh = h[base];
        phi += om * isc;
        float c  = __cosf(phi);
        float si = __sinf(phi);
        float wv = mg * hh;
        mr  += wv * c;
        mi  += wv * si;
        acc += mg;
        float inv  = 1.0f / (acc + 1e-8f);
        float memr = mr * inv;
        float memi = mi * inv;
        float rr = memr * c + memi * si;
        float ri = memi * c - memr * si;
        float hc = hh * c;
        float hs = hh * si;
        size_t cb = ((size_t)(b * SEQ + s)) * D4 + e;
        ctx[cb]            = hc;
        ctx[cb + DIM]      = hs;
        ctx[cb + 2 * DIM]  = rr;
        ctx[cb + 3 * DIM]  = ri;
        ctxbf[cb]           = f2bf(hc);
        ctxbf[cb + DIM]     = f2bf(hs);
        ctxbf[cb + 2 * DIM] = f2bf(rr);
        ctxbf[cb + 3 * DIM] = f2bf(ri);
    }
}

// ---------------- async staging of one 128x32 A-tile + 128x32 B-tile ----------------
__device__ __forceinline__ void stage_tiles(const bf16* A, const bf16* Bt, int K,
                                            int m0, int n0, int kb, unsigned bufbase,
                                            int r0, int s0, int r1, int s1) {
    const bf16* ga0 = A  + (size_t)(m0 + r0) * K + kb + s0 * 8;
    const bf16* ga1 = A  + (size_t)(m0 + r1) * K + kb + s1 * 8;
    const bf16* gb0 = Bt + (size_t)(n0 + r0) * K + kb + s0 * 8;
    const bf16* gb1 = Bt + (size_t)(n0 + r1) * K + kb + s1 * 8;
    unsigned la0 = bufbase + (unsigned)(r0 * LDS_STRIDE + s0 * 16);
    unsigned la1 = bufbase + (unsigned)(r1 * LDS_STRIDE + s1 * 16);
    unsigned lb0 = bufbase + LDS_A_BYTES + (unsigned)(r0 * LDS_STRIDE + s0 * 16);
    unsigned lb1 = bufbase + LDS_A_BYTES + (unsigned)(r1 * LDS_STRIDE + s1 * 16);
    asm volatile(
        "global_load_async_to_lds_b128 %0, %4, off\n\t"
        "global_load_async_to_lds_b128 %1, %5, off\n\t"
        "global_load_async_to_lds_b128 %2, %6, off\n\t"
        "global_load_async_to_lds_b128 %3, %7, off"
        :: "v"(la0), "v"(la1), "v"(lb0), "v"(lb1),
           "v"(ga0), "v"(ga1), "v"(gb0), "v"(gb1)
        : "memory");
}

// ---------------- LDS-tiled, double-buffered bf16 WMMA GEMM ----------------
// C[M,N] = A[M,K] * Bt[N,K]^T (+bias, epilogue)
// epi: 0 = bias store, 1 = sigmoid(v)*5 store, 2 = C += v (residual accumulate)
// Block = 256 threads (8 waves) computes a 128x128 C tile. Each wave computes a
// 32x64 sub-tile (2 A-frags x 4 B-frags -> 8 WMMAs, full fragment reuse).
// A/B tiles staged via gfx1250 async copies; double-buffered so the wait
// (s_wait_asynccnt 0x4, in-order completion) overlaps a full tile of compute.
__global__ __launch_bounds__(256)
void gemm_bf16_wmma_tiled(const bf16* __restrict__ A,
                          const bf16* __restrict__ Bt,
                          const float* __restrict__ bias,
                          float* __restrict__ C,
                          int M, int N, int K, int epi) {
    extern __shared__ char smem[];   // dynamic LDS: offset of smem[i] is i

    const int tid  = threadIdx.x;
    const int wave = tid >> 5;
    const int lane = tid & 31;
    const int lanemod = lane & 15;
    const int khi     = lane >> 4;

    const int nt = N / BN;
    const int m0 = (blockIdx.x / nt) * BM;
    const int n0 = (blockIdx.x % nt) * BN;

    const int wrow = (wave & 3) * 32;   // wave's row group within the 128-row tile
    const int wcol = (wave >> 2) * 64;  // wave's col group within the 128-col tile

    // each thread moves two 16B chunks of A-tile and two of B-tile per k-step
    const int r0 = tid >> 2,        s0 = tid & 3;          // chunk ids 0..255
    const int r1 = (tid + 256) >> 2, s1 = (tid + 256) & 3; // chunk ids 256..511

    v8f acc[8];
#pragma unroll
    for (int t = 0; t < 8; ++t) acc[t] = (v8f)(0.0f);

    // prologue: stage first K-tile into buffer 0
    stage_tiles(A, Bt, K, m0, n0, 0, 0u, r0, s0, r1, s1);

    unsigned buf = 0;
    for (int kb = 0; kb < K; kb += BK) {
        const int nxt = kb + BK;
        if (nxt < K) {
            stage_tiles(A, Bt, K, m0, n0, nxt, (buf ^ 1u) * LDS_HALF, r0, s0, r1, s1);
            asm volatile("s_wait_asynccnt 0x4" ::: "memory"); // current tile landed
        } else {
            asm volatile("s_wait_asynccnt 0x0" ::: "memory");
        }
        __syncthreads();

        const char* base  = smem + buf * LDS_HALF;
        const char* bbase = base + LDS_A_BYTES;

        // ---- 2 A fragments (rows wrow+ai*16+lanemod; K chunks at 8*khi, 16+8*khi) ----
        v16bf afrag[2];
#pragma unroll
        for (int ai = 0; ai < 2; ++ai) {
            const char* ap = base + (wrow + ai * 16 + lanemod) * LDS_STRIDE;
            v8bf alo = *(const v8bf*)(ap + 16 * khi);
            v8bf ahi = *(const v8bf*)(ap + 32 + 16 * khi);
#pragma unroll
            for (int j = 0; j < 8; ++j) { afrag[ai][j] = alo[j]; afrag[ai][8 + j] = ahi[j]; }
        }
        // ---- 4 B fragments (cols wcol+t*16+lanemod; K chunks at 8*khi, 16+8*khi) ----
        v16bf bfrag[4];
#pragma unroll
        for (int t = 0; t < 4; ++t) {
            const char* bp = bbase + (wcol + t * 16 + lanemod) * LDS_STRIDE;
            v8bf blo = *(const v8bf*)(bp + 16 * khi);
            v8bf bhi = *(const v8bf*)(bp + 32 + 16 * khi);
#pragma unroll
            for (int j = 0; j < 8; ++j) { bfrag[t][j] = blo[j]; bfrag[t][8 + j] = bhi[j]; }
        }
        // ---- 8 WMMAs: full 2x4 cross product ----
#pragma unroll
        for (int ai = 0; ai < 2; ++ai)
#pragma unroll
            for (int t = 0; t < 4; ++t)
                acc[ai * 4 + t] = __builtin_amdgcn_wmma_f32_16x16x32_bf16(
                    false, afrag[ai], false, bfrag[t], (short)0, acc[ai * 4 + t],
                    false, false);

        __syncthreads();   // reads done before next iteration overwrites this buffer
        buf ^= 1u;
    }

    // ---- epilogue: acc[ai*4+t][r] -> m = m0+wrow+ai*16+8*khi+r, n = n0+wcol+t*16+lanemod ----
#pragma unroll
    for (int ai = 0; ai < 2; ++ai) {
        const int mbase = m0 + wrow + ai * 16 + 8 * khi;
#pragma unroll
        for (int t = 0; t < 4; ++t) {
            int col = n0 + wcol + t * 16 + lanemod;
            float bv = bias[col];
#pragma unroll
            for (int r = 0; r < 8; ++r) {
                float v = acc[ai * 4 + t][r] + bv;
                if (epi == 1) v = 5.0f / (1.0f + __expf(-v));
                size_t idx = (size_t)(mbase + r) * N + col;
                if (epi == 2) C[idx] += v;
                else          C[idx] = v;
            }
        }
    }
}

// ---------------- host launcher ----------------
extern "C" void kernel_launch(void* const* d_in, const int* in_sizes, int n_in,
                              void* d_out, int out_size, void* d_ws, size_t ws_size,
                              hipStream_t stream) {
    const int*   tokens   = (const int*)  d_in[0];
    const float* emb      = (const float*)d_in[1];
    const float* ln_w     = (const float*)d_in[2];
    const float* ln_b     = (const float*)d_in[3];
    const float* W_omega  = (const float*)d_in[4];
    const float* b_omega  = (const float*)d_in[5];
    const float* int_sc   = (const float*)d_in[6];
    const float* W_mag    = (const float*)d_in[7];
    const float* b_mag    = (const float*)d_in[8];
    const float* W_out    = (const float*)d_in[9];
    const float* b_out    = (const float*)d_in[10];
    const float* fln_w    = (const float*)d_in[11];
    const float* fln_b    = (const float*)d_in[12];
    const float* W_head   = (const float*)d_in[13];
    const float* b_head   = (const float*)d_in[14];
    float* logits = (float*)d_out;

    char* ws = (char*)d_ws;
    size_t off = 0;
    auto alloc = [&](size_t bytes) { size_t o = off; off += (bytes + 255) & ~(size_t)255; return o; };
    float* x     = (float*)(ws + alloc((size_t)MROWS * DIM * 4));
    float* h     = (float*)(ws + alloc((size_t)MROWS * DIM * 4));
    bf16*  hbf   = (bf16*) (ws + alloc((size_t)MROWS * DIM * 2));
    float* omg   = (float*)(ws + alloc((size_t)MROWS * DIM * 4));
    float* mag   = (float*)(ws + alloc((size_t)MROWS * DIM * 4));
    float* ctx   = (float*)(ws + alloc((size_t)MROWS * D4 * 4));
    bf16*  ctxbf = (bf16*) (ws + alloc((size_t)MROWS * D4 * 2));
    bf16*  wT    = (bf16*) (ws + alloc((size_t)VOCAB * DIM * 2)); // largest transposed weight

    // 1) embedding + PE
    {
        int total = MROWS * DIM;
        embed_kernel<<<(total + 255) / 256, 256, 0, stream>>>(tokens, emb, x);
    }

    for (int l = 0; l < LAYERS; ++l) {
        layernorm_kernel<<<MROWS, 256, 0, stream>>>(x, ln_w + l * DIM, ln_b + l * DIM, h, hbf);

        // omega = h @ W_omega + b_omega
        {
            int total = DIM * DIM;
            transpose_convert_kernel<<<(total + 255) / 256, 256, 0, stream>>>(
                W_omega + (size_t)l * DIM * DIM, wT, DIM, DIM);
            int blocks = (MROWS / BM) * (DIM / BN);
            gemm_bf16_wmma_tiled<<<blocks, 256, LDS_TOTAL, stream>>>(
                hbf, wT, b_omega + l * DIM, omg, MROWS, DIM, DIM, 0);
        }
        // mag = sigmoid(h @ W_mag + b_mag) * 5
        {
            int total = DIM * DIM;
            transpose_convert_kernel<<<(total + 255) / 256, 256, 0, stream>>>(
                W_mag + (size_t)l * DIM * DIM, wT, DIM, DIM);
            int blocks = (MROWS / BM) * (DIM / BN);
            gemm_bf16_wmma_tiled<<<blocks, 256, LDS_TOTAL, stream>>>(
                hbf, wT, b_mag + l * DIM, mag, MROWS, DIM, DIM, 1);
        }
        // phase-interference scan -> ctx [M, 4D]
        tpi_scan_kernel<<<(BATCH * DIM + 255) / 256, 256, 0, stream>>>(
            h, omg, mag, int_sc + l * DIM, ctx, ctxbf);
        // x += ctx @ W_out + b_out
        {
            int total = D4 * DIM;
            transpose_convert_kernel<<<(total + 255) / 256, 256, 0, stream>>>(
                W_out + (size_t)l * D4 * DIM, wT, D4, DIM);
            int blocks = (MROWS / BM) * (DIM / BN);
            gemm_bf16_wmma_tiled<<<blocks, 256, LDS_TOTAL, stream>>>(
                ctxbf, wT, b_out + l * DIM, x, MROWS, DIM, D4, 2);
        }
    }

    // final LN
    layernorm_kernel<<<MROWS, 256, 0, stream>>>(x, fln_w, fln_b, h, hbf);

    // logits = h @ W_head + b_head
    {
        int total = DIM * VOCAB;
        transpose_convert_kernel<<<(total + 255) / 256, 256, 0, stream>>>(
            W_head, wT, DIM, VOCAB);
        int blocks = (MROWS / BM) * (VOCAB / BN);
        gemm_bf16_wmma_tiled<<<blocks, 256, LDS_TOTAL, stream>>>(
            hbf, wT, b_head, logits, MROWS, VOCAB, DIM, 0);
    }
    (void)in_sizes; (void)n_in; (void)out_size; (void)ws_size;
}
